// CausalAttention_84396107366730
// MI455X (gfx1250) — matrix operand order repaired and checked by reading, not compiled, not run on hardware
//
#include <hip/hip_runtime.h>
#include <hip/hip_bf16.h>

// ---------------------------------------------------------------------------
// CDNA5 (gfx1250) causal GQA attention, bf16 WMMA pipeline, wave32.
// ---------------------------------------------------------------------------

typedef __attribute__((ext_vector_type(16))) __bf16 v16bf;
typedef __attribute__((ext_vector_type(8)))  __bf16 v8bf;
typedef __attribute__((ext_vector_type(8)))  float  v8f;

#define D_  1024
#define H_  16
#define G_  4
#define HD_ 64
#define T_  4096
#define NQKV_ 1536   // H*HD + 2*G*HD

__device__ inline v16bf load2x8(const __bf16* p0, const __bf16* p1) {
    v8bf lo = *(const v8bf*)p0;
    v8bf hi = *(const v8bf*)p1;
    v16bf r;
#pragma unroll
    for (int i = 0; i < 8; ++i) { r[i] = lo[i]; r[i + 8] = hi[i]; }
    return r;
}

__device__ inline v8f wmma_bf16(v16bf a, v16bf b, v8f c) {
    return __builtin_amdgcn_wmma_f32_16x16x32_bf16(
        /*neg_a=*/false, a, /*neg_b=*/false, b,
        /*c_mod=*/(short)0, c, /*reuse_a=*/false, /*reuse_b=*/false);
}

// --------------------------- conversion kernels ----------------------------

__global__ void convert_f32_to_bf16(const float* __restrict__ src,
                                    __bf16* __restrict__ dst, int n) {
    int i = blockIdx.x * blockDim.x + threadIdx.x;
    if (i < n) dst[i] = (__bf16)src[i];
}

// src is K x N row-major; dst is N x K row-major (transposed) bf16.
__global__ void transpose_to_bf16(const float* __restrict__ src,
                                  __bf16* __restrict__ dst, int K, int N) {
    int i = blockIdx.x * blockDim.x + threadIdx.x;
    if (i >= K * N) return;
    int n = i % N, k = i / N;
    dst[(size_t)n * K + k] = (__bf16)src[i];
}

// ------------------------------ WMMA GEMM ----------------------------------
// C(MxN, f32, row-major) = A(MxK, bf16, row-major) * B(KxN), with B provided
// transposed (Bt is N x K row-major) so B-fragment loads are contiguous.
// One wave (32 threads) computes a 32x64 C tile: 2 A frags x 4 B frags ->
// 8 WMMAs per 32-wide k-step (~21 FLOP per byte of L2 fragment traffic).

__global__ void __launch_bounds__(32)
gemm_bf16_wmma(const __bf16* __restrict__ A, const __bf16* __restrict__ Bt,
               float* __restrict__ C, int M, int N, int K) {
    const int tn   = blockIdx.x;          // N / 64
    const int tm   = blockIdx.y;          // M / 32
    const int lane = threadIdx.x & 31;
    const int hl   = lane >> 4;           // lane half (0/1)
    const int l16  = lane & 15;

    const __bf16* ar0 = A + ((size_t)tm * 32 + l16) * K;       // rows l16, l16+16
    const __bf16* ar1 = ar0 + (size_t)16 * K;
    const __bf16* br[4];
#pragma unroll
    for (int j = 0; j < 4; ++j)
        br[j] = Bt + ((size_t)tn * 64 + j * 16 + l16) * K;     // cols of C

    v8f acc[2][4];
#pragma unroll
    for (int i = 0; i < 2; ++i)
#pragma unroll
        for (int j = 0; j < 4; ++j) { v8f z = {}; acc[i][j] = z; }

    for (int k0 = 0; k0 < K; k0 += 32) {
        // pull the next cachelines of each stream toward the WGP
        __builtin_prefetch((const void*)(ar0 + k0 + 256), 0, 3);
        __builtin_prefetch((const void*)(ar1 + k0 + 256), 0, 3);
#pragma unroll
        for (int j = 0; j < 4; ++j)
            __builtin_prefetch((const void*)(br[j] + k0 + 256), 0, 3);

        // A frag: k = k0 + (e/8)*16 + hl*8 + e%8
        v16bf a0 = load2x8(ar0 + k0 + hl * 8, ar0 + k0 + 16 + hl * 8);
        v16bf a1 = load2x8(ar1 + k0 + hl * 8, ar1 + k0 + 16 + hl * 8);
#pragma unroll
        for (int j = 0; j < 4; ++j) {
            // B frag: k = k0 + (e/8)*16 + hl*8 + e%8
            v16bf b = load2x8(br[j] + k0 + hl * 8, br[j] + k0 + 16 + hl * 8);
            acc[0][j] = wmma_bf16(a0, b, acc[0][j]);
            acc[1][j] = wmma_bf16(a1, b, acc[1][j]);
        }
    }
    // C layout: VGPR r -> row (+ r + hl*8), col (+ l16)
#pragma unroll
    for (int i = 0; i < 2; ++i)
#pragma unroll
        for (int j = 0; j < 4; ++j) {
            float* cp = C + ((size_t)tm * 32 + i * 16 + hl * 8) * N
                          + (size_t)tn * 64 + j * 16 + l16;
#pragma unroll
            for (int r = 0; r < 8; ++r) cp[(size_t)r * N] = acc[i][j][r];
        }
}

// --------------------- L2 norm + RoPE + layout shuffle ---------------------
// grid = (T, H + 2G), block = 32.  units 0..15: q heads, 16..19: k heads,
// 20..23: v heads (convert only).  Lane i owns the RoPE pair (i, i+32).

__global__ void __launch_bounds__(32)
normrope_kernel(const float* __restrict__ qkv,   // T x 1536
                __bf16* __restrict__ qh,         // [H][T][HD]
                __bf16* __restrict__ kh,         // [G][T][HD]
                __bf16* __restrict__ vt) {       // [G][HD][T]
    const int t    = blockIdx.x;
    const int unit = blockIdx.y;
    const int lane = threadIdx.x & 31;
    const float* row = qkv + (size_t)t * NQKV_;

    if (unit < H_ + G_) {
        const int off = (unit < H_) ? unit * HD_
                                    : H_ * HD_ + (unit - H_) * HD_;
        float x1 = row[off + lane];
        float x2 = row[off + 32 + lane];
        float ss = x1 * x1 + x2 * x2;
#pragma unroll
        for (int m = 16; m > 0; m >>= 1) ss += __shfl_xor(ss, m, 32);
        const float inv = 1.0f / (sqrtf(ss) + 1e-10f);
        x1 *= inv; x2 *= inv;
        // inv_freq = 10000^(-lane/32) = exp(-lane * ln(10000)/32)
        const float freq = __expf(-(float)lane * 0.28782313662426f);
        const float ang  = (float)t * freq;
        float s, c;
        __sincosf(ang, &s, &c);
        const float o1 = x1 * c - x2 * s;
        const float o2 = x2 * c + x1 * s;
        if (unit < H_) {
            __bf16* dst = qh + ((size_t)unit * T_ + t) * HD_;
            dst[lane] = (__bf16)o1; dst[lane + 32] = (__bf16)o2;
        } else {
            __bf16* dst = kh + ((size_t)(unit - H_) * T_ + t) * HD_;
            dst[lane] = (__bf16)o1; dst[lane + 32] = (__bf16)o2;
        }
    } else {
        const int g   = unit - H_ - G_;
        const int off = (H_ + G_) * HD_ + g * HD_;
        vt[((size_t)g * HD_ + lane) * T_ + t]      = (__bf16)row[off + lane];
        vt[((size_t)g * HD_ + lane + 32) * T_ + t] = (__bf16)row[off + 32 + lane];
    }
}

// --------------------------- flash attention -------------------------------
// One wave per (16-query tile, head).  Key blocks of 32; online softmax.

__global__ void __launch_bounds__(32)
flash_attn_kernel(const __bf16* __restrict__ qh,   // [H][T][HD]
                  const __bf16* __restrict__ kh,   // [G][T][HD]
                  const __bf16* __restrict__ vt,   // [G][HD][T]
                  __bf16* __restrict__ ao) {       // [T][H*HD]
    const int qt   = blockIdx.x;        // query tile (T/16)
    const int h    = blockIdx.y;
    const int g    = h >> 2;            // HPG = 4
    const int lane = threadIdx.x & 31;
    const int hl   = lane >> 4;
    const int l16  = lane & 15;

    __shared__ __align__(16) __bf16 pld[16 * 32];   // P tile, [m][k]

    // Q tile as two A fragments (d = 0..31 and 32..63)
    const __bf16* qrow = qh + ((size_t)h * T_ + (size_t)qt * 16 + l16) * HD_;
    const v16bf qa0 = load2x8(qrow +  0 + hl * 16, qrow +  8 + hl * 16);
    const v16bf qa1 = load2x8(qrow + 32 + hl * 16, qrow + 40 + hl * 16);

    v8f o0 = {}, o1 = {}, o2 = {}, o3 = {};
    float m_i[8], l_i[8];
#pragma unroll
    for (int r = 0; r < 8; ++r) { m_i[r] = -3.0e38f; l_i[r] = 0.0f; }

    const int qmax = qt * 16 + 15;
    const int nkb  = qmax / 32 + 1;
    const float scale = 0.125f;         // 1/sqrt(64)

    for (int kb = 0; kb < nkb; ++kb) {
        const int kbase = kb * 32;
        // ---- scores: two 16x16 C tiles (keys kbase..+15, kbase+16..+31)
        v8f s0 = {}, s1 = {};
        {
            const __bf16* kr0 = kh + ((size_t)g * T_ + kbase + l16) * HD_ + hl * 16;
            s0 = wmma_bf16(qa0, load2x8(kr0,      kr0 +  8), s0);
            s0 = wmma_bf16(qa1, load2x8(kr0 + 32, kr0 + 40), s0);
            const __bf16* kr1 = kr0 + 16 * HD_;
            s1 = wmma_bf16(qa0, load2x8(kr1,      kr1 +  8), s1);
            s1 = wmma_bf16(qa1, load2x8(kr1 + 32, kr1 + 40), s1);
        }
        // ---- masked online softmax (rows m = r + hl*8, cols = lanes of half)
#pragma unroll
        for (int r = 0; r < 8; ++r) {
            const int qi = qt * 16 + hl * 8 + r;
            float a = s0[r] * scale; if (kbase + l16      > qi) a = -1e30f;
            float b = s1[r] * scale; if (kbase + 16 + l16 > qi) b = -1e30f;
            float mx = fmaxf(a, b);
#pragma unroll
            for (int m = 8; m > 0; m >>= 1) mx = fmaxf(mx, __shfl_xor(mx, m, 32));
            const float mnew = fmaxf(m_i[r], mx);
            const float p0 = __expf(a - mnew);
            const float p1 = __expf(b - mnew);
            float rs = p0 + p1;
#pragma unroll
            for (int m = 8; m > 0; m >>= 1) rs += __shfl_xor(rs, m, 32);
            const float corr = __expf(m_i[r] - mnew);
            m_i[r] = mnew;
            l_i[r] = l_i[r] * corr + rs;
            o0[r] *= corr; o1[r] *= corr; o2[r] *= corr; o3[r] *= corr;
            s0[r] = p0; s1[r] = p1;
        }
        // ---- transpose P (C layout -> A layout) through LDS
#pragma unroll
        for (int r = 0; r < 8; ++r) {
            const int m = r + hl * 8;
            pld[m * 32 + l16]      = (__bf16)s0[r];
            pld[m * 32 + 16 + l16] = (__bf16)s1[r];
        }
        __syncthreads();
        const __bf16* pr = &pld[l16 * 32];
        const v16bf pa = load2x8(pr + hl * 16, pr + hl * 16 + 8);
        // ---- o += P @ V   (B frag: col d = j*16 + l16, k = key contiguous)
        const __bf16* vb = vt + (size_t)g * HD_ * T_ + kbase + hl * 16;
        {
            const __bf16* p0 = vb + (size_t)(0 * 16 + l16) * T_;
            o0 = wmma_bf16(pa, load2x8(p0, p0 + 8), o0);
            const __bf16* p1 = vb + (size_t)(1 * 16 + l16) * T_;
            o1 = wmma_bf16(pa, load2x8(p1, p1 + 8), o1);
            const __bf16* p2 = vb + (size_t)(2 * 16 + l16) * T_;
            o2 = wmma_bf16(pa, load2x8(p2, p2 + 8), o2);
            const __bf16* p3 = vb + (size_t)(3 * 16 + l16) * T_;
            o3 = wmma_bf16(pa, load2x8(p3, p3 + 8), o3);
        }
        __syncthreads();
    }
    // ---- write normalized output tile, bf16, [T][H*HD]
#pragma unroll
    for (int r = 0; r < 8; ++r) {
        const float inv = 1.0f / l_i[r];
        const size_t q = (size_t)qt * 16 + r + hl * 8;
        __bf16* op = ao + q * (H_ * HD_) + (size_t)h * HD_ + l16;
        op[0]  = (__bf16)(o0[r] * inv);
        op[16] = (__bf16)(o1[r] * inv);
        op[32] = (__bf16)(o2[r] * inv);
        op[48] = (__bf16)(o3[r] * inv);
    }
}

// ------------------------------- launcher ----------------------------------

extern "C" void kernel_launch(void* const* d_in, const int* in_sizes, int n_in,
                              void* d_out, int out_size, void* d_ws, size_t ws_size,
                              hipStream_t stream) {
    (void)in_sizes; (void)n_in; (void)out_size; (void)ws_size;
    const float* x     = (const float*)d_in[0];   // (1,4096,1024)
    const float* w_qkv = (const float*)d_in[1];   // (1024,1536)
    const float* w_o   = (const float*)d_in[2];   // (1024,1024)
    float* out = (float*)d_out;                   // (1,4096,1024)

    char* ws = (char*)d_ws;
    __bf16* xb  = (__bf16*)(ws + 0);              //  8 MB  x bf16
    __bf16* wqt = (__bf16*)(ws + 8388608);        //  3 MB  w_qkv^T bf16
    __bf16* wot = (__bf16*)(ws + 11534336);       //  2 MB  w_o^T bf16
    float*  qkv = (float*) (ws + 13631488);       // 24 MB  qkv f32
    __bf16* qhb = (__bf16*)(ws + 38797312);       //  8 MB  q [H][T][HD]
    __bf16* khb = (__bf16*)(ws + 47185920);       //  2 MB  k [G][T][HD]
    __bf16* vtb = (__bf16*)(ws + 49283072);       //  2 MB  v [G][HD][T]
    __bf16* aob = (__bf16*)(ws + 51380224);       //  8 MB  attn out bf16

    convert_f32_to_bf16<<<(T_ * D_ + 255) / 256, 256, 0, stream>>>(x, xb, T_ * D_);
    transpose_to_bf16<<<(D_ * NQKV_ + 255) / 256, 256, 0, stream>>>(w_qkv, wqt, D_, NQKV_);
    transpose_to_bf16<<<(D_ * D_ + 255) / 256, 256, 0, stream>>>(w_o, wot, D_, D_);

    gemm_bf16_wmma<<<dim3(NQKV_ / 64, T_ / 32), 32, 0, stream>>>(
        xb, wqt, qkv, T_, NQKV_, D_);

    normrope_kernel<<<dim3(T_, H_ + 2 * G_), 32, 0, stream>>>(qkv, qhb, khb, vtb);

    flash_attn_kernel<<<dim3(T_ / 16, H_), 32, 0, stream>>>(qhb, khb, vtb, aob);

    gemm_bf16_wmma<<<dim3(D_ / 64, T_ / 32), 32, 0, stream>>>(
        aob, wot, out, T_, D_, D_);
}
